// WindowAttention_52785148068378
// MI455X (gfx1250) — compile-verified, matrix operand order checked by reading, not run on hardware
//
#include <hip/hip_runtime.h>

typedef _Float16 h16;
typedef __attribute__((ext_vector_type(16))) _Float16 v16h;
typedef __attribute__((ext_vector_type(8)))  float    v8f;

// Problem constants
constexpr int NTOK = 64;   // tokens per window (4^3)
constexpr int CH   = 256;
constexpr int NH   = 8;
constexpr int HD   = 32;

// LDS layout (halfs). All fragment bases 16B-aligned; strides spread banks.
constexpr int XSTR  = 264;                 // [64][264] planes (x/attn, Q, K)
constexpr int VSTR  = 72;                  // vT rows  [head][32][72]
constexpr int PSTR  = 72;                  // P rows   [wave][16][72]
constexpr int LDS_PLANE = NTOK * XSTR;     // 16896 halfs
constexpr int LDS_VT = NH * HD * VSTR;     // 18432
constexpr int LDS_PB = NH * 16 * PSTR;     //  9216
constexpr int LDS_HALFS = 3 * LDS_PLANE + LDS_VT + LDS_PB;  // 78336
constexpr size_t LDS_BYTES = size_t(LDS_HALFS) * 2;          // 156672 B

// workspace layout
constexpr int WQ_HALFS  = 8 * 48 * 32 * 16;   // packed qkv_w (256x768) f16
constexpr int WP_HALFS  = 8 * 16 * 32 * 16;   // packed proj_w (256x256) f16
constexpr int RPB_FLOATS = NH * NTOK * NTOK;  // gathered rel-pos bias
constexpr size_t RPB_BYTE_OFF = size_t(WQ_HALFS + WP_HALFS) * 2;  // 524288

// ---- fragment helpers -------------------------------------------------------
// 16-bit A (or B-as-columns) fragment from LDS. p -> [row][col0], row already
// includes (lane&15). Fragment = two contiguous 16B runs: halfs [kb..kb+7] and
// [kb+16..kb+23] where kb = 8*laneHi  =>  2x ds_load_b128.
__device__ inline v16h lds_fragA(const h16* p, int laneHi) {
  const uint4* q = reinterpret_cast<const uint4*>(p + laneHi * 8);
  union { v16h v; uint4 x[2]; } r;
  r.x[0] = q[0];
  r.x[1] = q[2];   // +32 bytes
  return r.v;
}

// B fragment from pre-swizzled global weights: 32 contiguous bytes per lane.
__device__ inline v16h ldg_fragB(const h16* w) {
  union { v16h v; uint4 q[2]; } r;
  const uint4* p = reinterpret_cast<const uint4*>(w);
  r.q[0] = p[0];
  r.q[1] = p[1];
  return r.v;
}

__device__ inline v8f wmma16(v16h a, v16h b, v8f c) {
  return __builtin_amdgcn_wmma_f32_16x16x32_f16(false, a, false, b, (short)0, c,
                                                false, false);
}

// ---- pre-pack: weights f32->f16 in B-fragment order + rpb gather ------------
__global__ void pack_weights(const float* __restrict__ qkv_w,
                             const float* __restrict__ proj_w,
                             const float* __restrict__ bias_table,
                             h16* __restrict__ wq, h16* __restrict__ wp,
                             float* __restrict__ rpb) {
  int id = blockIdx.x * blockDim.x + threadIdx.x;
  if (id < WQ_HALFS) {
    int hh = id & 15, lane = (id >> 4) & 31, tile = id >> 9;
    int nT = tile % 48, kS = tile / 48;
    int k = kS * 32 + (hh & 7) + ((hh & 8) ? 16 : 0) + ((lane & 16) ? 8 : 0);
    int n = nT * 16 + (lane & 15);
    wq[id] = (h16)qkv_w[k * 768 + n];
  } else if (id < WQ_HALFS + WP_HALFS) {
    int j = id - WQ_HALFS;
    int hh = j & 15, lane = (j >> 4) & 31, tile = j >> 9;
    int nT = tile & 15, kS = tile >> 4;
    int k = kS * 32 + (hh & 7) + ((hh & 8) ? 16 : 0) + ((lane & 16) ? 8 : 0);
    int n = nT * 16 + (lane & 15);
    wp[j] = (h16)proj_w[k * 256 + n];
  } else if (id < WQ_HALFS + WP_HALFS + RPB_FLOATS) {
    int j = id - (WQ_HALFS + WP_HALFS);
    int h = j >> 12, rem = j & 4095;
    int tq = rem >> 6, tk = rem & 63;
    int zq = tq >> 4, yq = (tq >> 2) & 3, xq = tq & 3;
    int zk = tk >> 4, yk = (tk >> 2) & 3, xk = tk & 3;
    int idx = ((zq - zk + 3) * 7 + (yq - yk + 3)) * 7 + (xq - xk + 3);
    rpb[j] = bias_table[idx * 8 + h];
  }
}

// ---- fused window attention -------------------------------------------------
__global__ __launch_bounds__(256) void swin_win_attn(
    const float* __restrict__ x, const h16* __restrict__ wq,
    const float* __restrict__ qkv_b, const h16* __restrict__ wp,
    const float* __restrict__ proj_b, const float* __restrict__ rpb,
    float* __restrict__ out) {
  extern __shared__ char smem_raw[];
  h16* xh   = reinterpret_cast<h16*>(smem_raw);  // x tile, later attn-out tile
  h16* qbuf = xh + LDS_PLANE;                    // Q  [64][264], c = head*32+hd
  h16* kbuf = qbuf + LDS_PLANE;                  // K  [64][264]
  h16* vT   = kbuf + LDS_PLANE;                  // V^T [head][hd][token]
  h16* pb   = vT + LDS_VT;                       // per-wave P staging

  const int tid = threadIdx.x;
  const int lane = tid & 31, wave = tid >> 5;
  const int lm = lane & 15, laneHi = lane >> 4;
  const int widx = blockIdx.x;
  const int wd = widx >> 8, wh = (widx >> 4) & 15, ww = widx & 15;
  const v8f vzero = {};

  auto tok_base = [&](int t) -> long {
    int z = t >> 4, y = (t >> 2) & 3, xx = t & 3;
    long d = wd * 4 + z, hy = wh * 4 + y, wx = ww * 4 + xx;
    return ((d * 64 + hy) * 64 + wx) * 256;
  };

  // Phase 1: stage window (64x256) into LDS as f16. float4 loads, b64 stores.
  for (int i = tid; i < NTOK * CH / 4; i += 256) {
    int t = i >> 6, c4 = (i & 63) << 2;
    float4 v = *reinterpret_cast<const float4*>(x + tok_base(t) + c4);
    union { uint2 u; h16 e[4]; } p;
    p.e[0] = (h16)v.x; p.e[1] = (h16)v.y; p.e[2] = (h16)v.z; p.e[3] = (h16)v.w;
    *reinterpret_cast<uint2*>(xh + t * XSTR + c4) = p.u;
  }
  __syncthreads();

  // Phase 2: QKV = xh @ Wqkv + b   (64x256 @ 256x768), 6 N-tiles per wave.
  for (int nt = wave; nt < 48; nt += 8) {
    v16h b[8];
#pragma unroll
    for (int ks = 0; ks < 8; ++ks) {
      b[ks] = ldg_fragB(wq + (size_t)((ks * 48 + nt) * 32 + lane) * 16);
      if (nt + 8 < 48)
        __builtin_prefetch(wq + (size_t)((ks * 48 + nt + 8) * 32 + lane) * 16, 0, 0);
    }
    v8f acc[4] = {vzero, vzero, vzero, vzero};
#pragma unroll
    for (int ks = 0; ks < 8; ++ks) {
#pragma unroll
      for (int mt = 0; mt < 4; ++mt) {   // 4 independent WMMAs per K-step
        v16h a = lds_fragA(xh + (mt * 16 + lm) * XSTR + ks * 32, laneHi);
        acc[mt] = wmma16(a, b[ks], acc[mt]);
      }
    }
    // epilogue: wave-uniform routing (scalar branch via readfirstlane)
    const int ntU  = __builtin_amdgcn_readfirstlane(nt);
    const int which = ntU >> 4;
    const int head  = (ntU >> 1) & 7;
    const int hd    = ((ntU & 1) << 4) + lm;
    const float bias = qkv_b[ntU * 16 + lm];
    if (which < 2) {
      h16* dst = (which == 0 ? qbuf : kbuf) + head * 32 + hd;
#pragma unroll
      for (int mt = 0; mt < 4; ++mt)
#pragma unroll
        for (int r = 0; r < 8; ++r) {
          int t = mt * 16 + r + 8 * laneHi;
          dst[t * XSTR] = (h16)(acc[mt][r] + bias);
        }
    } else {
      h16* dst = vT + head * HD * VSTR + hd * VSTR;   // contiguous in t
#pragma unroll
      for (int mt = 0; mt < 4; ++mt)
#pragma unroll
        for (int r = 0; r < 8; r += 2) {
          int t = mt * 16 + r + 8 * laneHi;           // even
          union { unsigned u; h16 e[2]; } pk;
          pk.e[0] = (h16)(acc[mt][r] + bias);
          pk.e[1] = (h16)(acc[mt][r + 1] + bias);
          *reinterpret_cast<unsigned*>(dst + t) = pk.u;
        }
    }
  }
  __syncthreads();

  // Phase 3/4: one head per wave.
  const h16* vh = vT + wave * HD * VSTR;
  const float* rpb_h = rpb + wave * (NTOK * NTOK);
  h16* ph = pb + wave * 16 * PSTR;
  const float scale = 0.17677669529663689f;   // 1/sqrt(32)

  for (int mt = 0; mt < 4; ++mt) {
    // S = Q @ K^T  (16x64 slice), single K-step since hd == 32.
    v16h aq = lds_fragA(qbuf + (mt * 16 + lm) * XSTR + wave * 32, laneHi);
    v8f s[4];
#pragma unroll
    for (int nt = 0; nt < 4; ++nt) {
      v16h bk = lds_fragA(kbuf + (nt * 16 + lm) * XSTR + wave * 32, laneHi);
      s[nt] = wmma16(aq, bk, vzero);
    }
    // scale + precomputed rel-pos bias + softmax (rows split by laneHi groups)
#pragma unroll
    for (int r = 0; r < 8; ++r) {
      int tq = mt * 16 + r + 8 * laneHi;
      const float* brow = rpb_h + tq * 64 + lm;
      float m = -3.4e38f;
#pragma unroll
      for (int nt = 0; nt < 4; ++nt) {
        float val = s[nt][r] * scale + brow[nt * 16];
        s[nt][r] = val;
        m = fmaxf(m, val);
      }
#pragma unroll
      for (int mk = 1; mk < 16; mk <<= 1) m = fmaxf(m, __shfl_xor(m, mk, 16));
      float sum = 0.f;
#pragma unroll
      for (int nt = 0; nt < 4; ++nt) {
        float p = __expf(s[nt][r] - m);
        s[nt][r] = p;
        sum += p;
      }
#pragma unroll
      for (int mk = 1; mk < 16; mk <<= 1) sum += __shfl_xor(sum, mk, 16);
      float inv = 1.f / sum;
#pragma unroll
      for (int nt = 0; nt < 4; ++nt)
        ph[(r + 8 * laneHi) * PSTR + nt * 16 + lm] = (h16)(s[nt][r] * inv);
    }
    // O = P @ V -> [16 x 32]; two independent accumulators across nt2.
    v8f o[2] = {vzero, vzero};
#pragma unroll
    for (int ks = 0; ks < 2; ++ks) {
      v16h ap = lds_fragA(ph + lm * PSTR + ks * 32, laneHi);
#pragma unroll
      for (int nt2 = 0; nt2 < 2; ++nt2) {
        v16h bv = lds_fragA(vh + (nt2 * 16 + lm) * VSTR + ks * 32, laneHi);
        o[nt2] = wmma16(ap, bv, o[nt2]);
      }
    }
#pragma unroll
    for (int nt2 = 0; nt2 < 2; ++nt2)
#pragma unroll
      for (int r = 0; r < 8; ++r)
        xh[(mt * 16 + r + 8 * laneHi) * XSTR + wave * 32 + nt2 * 16 + lm] =
            (h16)o[nt2][r];
  }
  __syncthreads();

  // Phase 5: out = attn @ Wproj + b. Stage fp32 tile in LDS (reuses Q/K
  // planes), then fully-coalesced float4 global stores.
  float* outs = reinterpret_cast<float*>(qbuf);   // 64 x 256 f32 = 64KB
  for (int nt = wave * 2; nt < wave * 2 + 2; ++nt) {
    v16h b[8];
#pragma unroll
    for (int ks = 0; ks < 8; ++ks)
      b[ks] = ldg_fragB(wp + (size_t)((ks * 16 + nt) * 32 + lane) * 16);
    v8f acc[4] = {vzero, vzero, vzero, vzero};
#pragma unroll
    for (int ks = 0; ks < 8; ++ks) {
#pragma unroll
      for (int mt = 0; mt < 4; ++mt) {
        v16h a = lds_fragA(xh + (mt * 16 + lm) * XSTR + ks * 32, laneHi);
        acc[mt] = wmma16(a, b[ks], acc[mt]);
      }
    }
    int c = nt * 16 + lm;
    float bias = proj_b[c];
#pragma unroll
    for (int mt = 0; mt < 4; ++mt)
#pragma unroll
      for (int r = 0; r < 8; ++r) {
        int t = mt * 16 + r + 8 * laneHi;
        outs[t * 256 + c] = acc[mt][r] + bias;
      }
  }
  __syncthreads();
  for (int i = tid; i < NTOK * CH / 4; i += 256) {
    int t = i >> 6, c4 = (i & 63) << 2;
    *reinterpret_cast<float4*>(out + tok_base(t) + c4) =
        *reinterpret_cast<const float4*>(outs + t * 256 + c4);
  }
}

extern "C" void kernel_launch(void* const* d_in, const int* in_sizes, int n_in,
                              void* d_out, int out_size, void* d_ws,
                              size_t ws_size, hipStream_t stream) {
  const float* x          = (const float*)d_in[0];
  const float* qkv_w      = (const float*)d_in[1];
  const float* qkv_b      = (const float*)d_in[2];
  const float* proj_w     = (const float*)d_in[3];
  const float* proj_b     = (const float*)d_in[4];
  const float* bias_table = (const float*)d_in[5];
  float* out = (float*)d_out;

  h16* wq = (h16*)d_ws;
  h16* wp = wq + WQ_HALFS;
  float* rpb = reinterpret_cast<float*>((char*)d_ws + RPB_BYTE_OFF);

  (void)hipFuncSetAttribute(reinterpret_cast<const void*>(swin_win_attn),
                            hipFuncAttributeMaxDynamicSharedMemorySize,
                            (int)LDS_BYTES);

  int packN = WQ_HALFS + WP_HALFS + RPB_FLOATS;   // 294912
  pack_weights<<<(packN + 255) / 256, 256, 0, stream>>>(qkv_w, proj_w,
                                                        bias_table, wq, wp, rpb);
  swin_win_attn<<<4096, 256, LDS_BYTES, stream>>>(x, wq, qkv_b, wp, proj_b,
                                                  rpb, out);
}
